// Filtered_NCC_vxm_29703993819922
// MI455X (gfx1250) — matrix operand order repaired and checked by reading, not compiled, not run on hardware
//
#include <hip/hip_runtime.h>
#include <cmath>

// CDNA5 / gfx1250: wave32, WMMA 16x16x32 f16->f32.
typedef __attribute__((ext_vector_type(16))) _Float16 v16h;
typedef __attribute__((ext_vector_type(8)))  float    v8f;

#define N160 160
#define NROW (N160 * N160)          // 25600 lines per conv pass
#define VOL  (N160 * N160 * N160)   // 4,096,000 voxels

struct Taps9 { float w[9]; };

__device__ __forceinline__ int reflect160(int i) {
  i = (i < 0) ? (-1 - i) : i;
  i = (i >= N160) ? (2 * N160 - 1 - i) : i;
  return i;
}

// ---------------------------------------------------------------------------
// Generic 9-tap 1D correlation along one axis of a 160^3 volume on the WMMA
// pipe:  D(16x16,f32) = A(16x32,f16) x B(32x16,f16) + 0.
//   M: 16 lines perpendicular to the conv axis (tile never crosses the outer
//      stride boundary since 160 % 16 == 0 -> one scalar base per tile)
//   N: 16 consecutive output positions along the conv axis
//   K: 24 input samples covering the 9-tap footprint (+8 zero rows in B)
//   B[k][n] = w[k-n] for 0<=k-n<=8 (constant banded tap matrix)
// One v_wmma_f32_16x16x32_f16 replaces 72 wave32 VALU FMAs per tile.
// Compile-time CS/OS/IS (conv/outer/inner strides), pad mode and optional
// element-wise pre-multiply -> branchless 32-bit addressing, immediate-offset
// stores, s_clause'd loads. in_scale is an exact power-of-2 range shift for
// f16 safety (compensated in the taps of the last pass of a chain).
// ---------------------------------------------------------------------------
template <int CS, int OS, int IS, bool ZERO, bool HAS2>
__global__ void __launch_bounds__(256)
conv9_wmma(const float* __restrict__ in, const float* __restrict__ in2,
           float* __restrict__ out, Taps9 taps, float in_scale)
{
  const int lane = threadIdx.x & 31;
  const int wave = blockIdx.x * 8 + (threadIdx.x >> 5);
  const int row_tile = wave / 10;            // 1600 row tiles
  const int j_tile   = wave % 10;            // 10 tiles along conv axis
  const int jb = j_tile * 16;

  // Wave-uniform tile base (32-bit; max offset < 2^23)
  const int sbase = (row_tile / 10) * OS + ((row_tile % 10) * 16) * IS;

  // ---- A: 16x32 f16. ISA layout: lane L holds row m=L&15;
  //      K(e) = ((L>>4)*8) + e + (e>=8 ? 8 : 0);  K<=23 <=> e<8 || lane<16.
  const int m     = lane & 15;
  const int khalf = (lane >> 4) << 3;
  const int rbase = sbase + m * IS;
  const bool hi8  = (lane < 16);             // lanes 0..15 own K=16..23

  v16h a;
  const bool interior = (jb >= 4) && (jb + 20 <= N160);  // j_tile in [1,8]
  if (interior) {
    // Purely affine addresses -> immediate offsets, no clamps.
    const int base0 = rbase + (jb + khalf - 4) * CS;     // K = khalf..khalf+7
    const int base1 = rbase + (jb + 12) * CS;            // K = 16..23 (<=159)
#pragma unroll
    for (int e = 0; e < 8; ++e) {
      float v = in[base0 + e * CS];
      if (HAS2) v *= in2[base0 + e * CS];
      a[e] = (_Float16)(v * in_scale);
    }
#pragma unroll
    for (int e = 0; e < 8; ++e) {
      float v = in[base1 + e * CS];
      if (HAS2) v *= in2[base1 + e * CS];
      a[8 + e] = (_Float16)(hi8 ? v * in_scale : 0.0f);
    }
  } else {
    // Boundary tiles: unconditional loads at clamped/reflected addresses,
    // out-of-window lanes zeroed by select (no per-element branches).
#pragma unroll
    for (int e = 0; e < 16; ++e) {
      const bool second = (e >= 8);
      int x = second ? (jb + 12 + (e - 8)) : (jb + khalf - 4 + e);
      bool ok = second ? hi8 : true;
      if (ZERO) {
        ok = ok && (x >= 0) && (x < N160);
        x = max(0, min(x, N160 - 1));
      } else {
        x = reflect160(x);                   // scipy 'reflect' == symmetric
      }
      float v = in[rbase + x * CS];
      if (HAS2) v *= in2[rbase + x * CS];
      a[e] = (_Float16)(ok ? v * in_scale : 0.0f);
    }
  }

  // gfx1250 global_prefetch_b8 for the next tile along the conv axis
  if (j_tile < 9)
    __builtin_prefetch(&in[rbase + (jb + 16) * CS], 0, 1);

  // ---- B: 32x16 banded tap matrix. Lane L holds column n=L&15,
  //      K(e) = ((L>>4)*8) + e + (e>=8 ? 8 : 0).
  const int n  = lane & 15;
  const int kb = (lane >> 4) << 3;
  v16h b;
#pragma unroll
  for (int e = 0; e < 16; ++e) {
    const int d = kb + e + (e >= 8 ? 8 : 0) - n;
    float tv = 0.0f;
#pragma unroll
    for (int t = 0; t < 9; ++t) tv = (d == t) ? taps.w[t] : tv;
    b[e] = (_Float16)tv;
  }

  v8f c = {};
  // (neg_a, A, neg_b, B, c_mod, C, reuse_a, reuse_b)
  c = __builtin_amdgcn_wmma_f32_16x16x32_f16(false, a, false, b, (short)0, c,
                                             false, false);

  // ---- D: 16x16 f32. Lane L holds col n=L&15, rows ((L>>4)*8)+v.
  const int mb = (lane >> 4) << 3;
  float* __restrict__ po = out + (sbase + mb * IS + (jb + n) * CS);
#pragma unroll
  for (int v = 0; v < 8; ++v) po[v * IS] = c[v];   // immediate offsets
}

// ---------------------------------------------------------------------------
// 3x3x3 spatial median, symmetric padding. 14 static bubble passes -> v[13]
// is the median of 27; all indices static so the array stays in VGPRs.
// ---------------------------------------------------------------------------
__global__ void __launch_bounds__(256)
median27_kernel(const float* __restrict__ in, float* __restrict__ out)
{
  const int i = blockIdx.x * blockDim.x + threadIdx.x;
  if (i >= VOL) return;
  const int x = i % N160;
  const int y = (i / N160) % N160;
  const int z = i / (N160 * N160);

  float v[27];
  int c = 0;
#pragma unroll
  for (int dz = -1; dz <= 1; ++dz)
#pragma unroll
    for (int dy = -1; dy <= 1; ++dy)
#pragma unroll
      for (int dx = -1; dx <= 1; ++dx) {
        const int zz = reflect160(z + dz);
        const int yy = reflect160(y + dy);
        const int xx = reflect160(x + dx);
        v[c++] = in[(zz * N160 + yy) * N160 + xx];
      }
#pragma unroll
  for (int p = 0; p < 14; ++p)
#pragma unroll
    for (int q = 26; q > p; --q) {
      const float lo = fminf(v[q - 1], v[q]);
      const float hi = fmaxf(v[q - 1], v[q]);
      v[q - 1] = lo; v[q] = hi;
    }
  out[i] = v[13];
}

__global__ void add_inplace_kernel(float* __restrict__ a,
                                   const float* __restrict__ b, int nel) {
  for (int i = blockIdx.x * blockDim.x + threadIdx.x; i < nel;
       i += gridDim.x * blockDim.x)
    a[i] += b[i];
}

__global__ void log_combine_kernel(const float* __restrict__ t1,
                                   const float* __restrict__ t2,
                                   const float* __restrict__ g, float k,
                                   float* __restrict__ out, int nel) {
  for (int i = blockIdx.x * blockDim.x + threadIdx.x; i < nel;
       i += gridDim.x * blockDim.x)
    out[i] = t1[i] + t2[i] + k * g[i];
}

// cc = cross^2 / (Ivar*Jvar + 1e-5), block tree reduce to partials[]
__global__ void __launch_bounds__(256)
cc_partial_kernel(const float* __restrict__ Is, const float* __restrict__ Js,
                  const float* __restrict__ I2s, const float* __restrict__ J2s,
                  const float* __restrict__ IJs, float* __restrict__ partials,
                  int nel)
{
  __shared__ float sm[256];
  const float wsz = 729.0f, inv = 1.0f / 729.0f;
  float local = 0.0f;
  for (int i = blockIdx.x * blockDim.x + threadIdx.x; i < nel;
       i += gridDim.x * blockDim.x) {
    const float S = Is[i], T = Js[i];
    const float uI = S * inv, uJ = T * inv;
    const float cross = IJs[i] - uJ * S - uI * T + uI * uJ * wsz;
    const float Iv = I2s[i] - 2.0f * uI * S + uI * uI * wsz;
    const float Jv = J2s[i] - 2.0f * uJ * T + uJ * uJ * wsz;
    local += cross * cross / (Iv * Jv + 1e-5f);
  }
  sm[threadIdx.x] = local; __syncthreads();
  for (int s = 128; s > 0; s >>= 1) {
    if (threadIdx.x < s) sm[threadIdx.x] += sm[threadIdx.x + s];
    __syncthreads();
  }
  if (threadIdx.x == 0) partials[blockIdx.x] = sm[0];
}

__global__ void acc_partials_kernel(const float* __restrict__ partials, int nb,
                                    float* __restrict__ acc) {
  __shared__ float sm[256];
  float local = 0.0f;
  for (int i = threadIdx.x; i < nb; i += blockDim.x) local += partials[i];
  sm[threadIdx.x] = local; __syncthreads();
  for (int s = 128; s > 0; s >>= 1) {
    if (threadIdx.x < s) sm[threadIdx.x] += sm[threadIdx.x + s];
    __syncthreads();
  }
  if (threadIdx.x == 0) acc[0] += sm[0];
}

__global__ void zero_kernel(float* p) { p[0] = 0.0f; }

__global__ void finalize_kernel(const float* acc, float* out) {
  out[0] = acc[0] * (-1.0f / (5.0f * (float)VOL));   // avg of 5 * (-mean)
}

// ---------------------------------------------------------------------------
extern "C" void kernel_launch(void* const* d_in, const int* in_sizes, int n_in,
                              void* d_out, int out_size, void* d_ws,
                              size_t ws_size, hipStream_t stream)
{
  (void)in_sizes; (void)n_in; (void)out_size; (void)ws_size;
  const float* yt = (const float*)d_in[0];
  const float* yp = (const float*)d_in[1];
  float* out = (float*)d_out;

  // Workspace: acc(256B) | partials(1024 f32) | 9 volumes (~148 MB, L2-sized)
  char* ws = (char*)d_ws;
  const size_t VB = (size_t)VOL * sizeof(float);
  float* acc      = (float*)ws;
  float* partials = (float*)(ws + 256);
  char*  volbase  = ws + 256 + 1024 * sizeof(float);
  float* B[9];
  for (int k = 0; k < 9; ++k) B[k] = (float*)(volbase + (size_t)k * VB);
  float* bufI = B[0]; float* bufJ = B[1];
  float* S0 = B[2]; float* S1 = B[3]; float* S2 = B[4];
  float* S3 = B[5]; float* S4 = B[6];
  float* t0 = B[7]; float* t1 = B[8];

  // Tap sets (host-side, deterministic).
  Taps9 g0t, g2t, sm121, dz16, ones1, ones4096;
  double s2sum = 0.0;
  {
    double phi[9], s = 0.0;
    for (int i = 0; i < 9; ++i) { double xx = i - 4; phi[i] = std::exp(-0.5 * xx * xx); s += phi[i]; }
    for (int i = 0; i < 9; ++i) {
      phi[i] /= s;
      const double xx = i - 4;
      g0t.w[i] = (float)phi[i];
      g2t.w[i] = (float)((xx * xx - 1.0) * phi[i]);
      s2sum += (xx * xx - 1.0) * phi[i];
    }
    for (int i = 0; i < 9; ++i) { sm121.w[i] = 0.f; dz16.w[i] = 0.f; ones1.w[i] = 1.f; ones4096.w[i] = 4096.f; }
    sm121.w[3] = 1.f; sm121.w[4] = 2.f; sm121.w[5] = 1.f;
    dz16.w[3] = -16.f; dz16.w[5] = 16.f;   // sobel: 16 * [-1,0,1] (B,C axes -> x4 each)
  }

  const dim3 G(2000), Bk(256);  // 16000 waves = 1600 row tiles x 10 j tiles

  // Symmetric-pad filter passes (z/y/x)
  auto convZ = [&](const float* i, float* o, const Taps9& t) {
    conv9_wmma<NROW, N160, 1, false, false><<<G, Bk, 0, stream>>>(i, nullptr, o, t, 1.0f);
  };
  auto convY = [&](const float* i, float* o, const Taps9& t) {
    conv9_wmma<N160, NROW, 1, false, false><<<G, Bk, 0, stream>>>(i, nullptr, o, t, 1.0f);
  };
  auto convX = [&](const float* i, float* o, const Taps9& t) {
    conv9_wmma<1, NROW, N160, false, false><<<G, Bk, 0, stream>>>(i, nullptr, o, t, 1.0f);
  };
  // Zero-pad box-sum passes: 1/16 range shift per pass, x4096 in final taps
  auto boxZ = [&](const float* i, const float* i2, float* o) {
    if (i2) conv9_wmma<NROW, N160, 1, true, true ><<<G, Bk, 0, stream>>>(i, i2, o, ones1, 0.0625f);
    else    conv9_wmma<NROW, N160, 1, true, false><<<G, Bk, 0, stream>>>(i, nullptr, o, ones1, 0.0625f);
  };
  auto boxY = [&](const float* i, float* o) {
    conv9_wmma<N160, NROW, 1, true, false><<<G, Bk, 0, stream>>>(i, nullptr, o, ones1, 0.0625f);
  };
  auto boxX = [&](const float* i, float* o) {
    conv9_wmma<1, NROW, N160, true, false><<<G, Bk, 0, stream>>>(i, nullptr, o, ones4096, 0.0625f);
  };

  zero_kernel<<<1, 1, 0, stream>>>(acc);

  for (int f = 0; f < 5; ++f) {
    for (int s = 0; s < 2; ++s) {
      const float* v = s ? yp : yt;
      float* dst = s ? bufJ : bufI;
      if (f == 0) {
        // identity: consume d_in directly below
      } else if (f == 1) {             // 3D Gaussian
        convZ(v, t0, g0t); convY(t0, t1, g0t); convX(t1, dst, g0t);
      } else if (f == 2) {             // Sobel (only z-term survives) * 16
        convX(v, t0, sm121); convY(t0, t1, sm121); convZ(t1, dst, dz16);
      } else if (f == 3) {             // LoG = sum g2-terms + 2*(sum g2)*gauss
        convZ(v, t0, g0t);             // gz0
        convZ(v, t1, g2t);             // gz2
        convY(t0, S0, g0t);            // P = g0y(gz0)
        convY(t0, S1, g2t);            // Q = g2y(gz0)
        convY(t1, S2, g0t);            // R = g0y(gz2)
        add_inplace_kernel<<<4096, 256, 0, stream>>>(S1, S2, VOL);  // Q+R
        convX(S1, t0, g0t);            // g0x(Q+R)
        convX(S0, t1, g2t);            // g2x(P)
        convX(S0, S2, g0t);            // gauss3d
        log_combine_kernel<<<4096, 256, 0, stream>>>(
            t0, t1, S2, (float)(2.0 * s2sum), dst, VOL);
      } else {                         // 3^3 median
        median27_kernel<<<(VOL + 255) / 256, 256, 0, stream>>>(v, dst);
      }
    }
    const float* fI = (f == 0) ? yt : (const float*)bufI;
    const float* fJ = (f == 0) ? yp : (const float*)bufJ;

    // NCC box sums (zero pad), premultiply fused into the z-pass.
    const float* qa[5] = {fI, fJ, fI, fJ, fI};
    const float* qb[5] = {nullptr, nullptr, fI, fJ, fJ};
    float* qs[5] = {S0, S1, S2, S3, S4};
    for (int q = 0; q < 5; ++q) {
      boxZ(qa[q], qb[q], t0);
      boxY(t0, t1);
      boxX(t1, qs[q]);
    }
    cc_partial_kernel<<<1024, 256, 0, stream>>>(S0, S1, S2, S3, S4, partials, VOL);
    acc_partials_kernel<<<1, 256, 0, stream>>>(partials, 1024, acc);
  }
  finalize_kernel<<<1, 1, 0, stream>>>(acc, out);
}
